// lpLSTM_13606456393976
// MI455X (gfx1250) — matrix operand order repaired and checked by reading, not compiled, not run on hardware
//
#include <hip/hip_runtime.h>

// ---------------------------------------------------------------------------
// Low-pass LSTM for MI455X (gfx1250), wave32 + v_wmma_f32_16x16x32_bf16.
//
// Fused recurrence: per time step t, each wave computes a 32(batch) x 16(H)
// output tile for ALL four gates:
//   gates[4] = bias + [h_t ; x_t] (64 x 2048, bf16)  @  [W_h ; W_x] (2048 x 4096, bf16)
// then applies sigmoid/tanh, the cell update, and the retention blend, writing
// h_{t+1} both as fp32 (output sequence) and bf16 (next step's A matrix).
//
// Each wave owns TWO batch-tiles so every B fragment (weights) feeds two
// WMMAs: this halves the per-step weight stream from L2 (the binding resource
// for this tiny-M GEMM) while keeping 128 waves of parallelism.
//
// Weights are packed once into the exact CDNA5 WMMA B-fragment layout
// (ISA 7.12.2, 16-bit B 32x16: lane L -> col L&15, k = (L<16?0:8)+e+(e>=8?8:0)), so each
// lane reads its whole fragment as one contiguous 32-byte chunk. A-fragments
// come straight from row-major storage (lane L -> row L&15, k runs
// {base..base+7, base+16..base+23}, base = L<16?0:8) as two 16-byte loads.
// ---------------------------------------------------------------------------

typedef __attribute__((ext_vector_type(16))) __bf16 v16bf;
typedef __attribute__((ext_vector_type(8)))  __bf16 v8bf;
typedef __attribute__((ext_vector_type(8)))  float  v8f;

#define T_STEPS 512
#define BATCH   64
#define HDIM    1024
#define BH      (BATCH * HDIM)          // 65536 elements of state
#define KT64    64                      // k-tiles per gate: 32 from W_h + 32 from W_x
#define NT      64                      // n-tiles: HDIM / 16
#define TILE_ELEMS (32 * 16)            // 512 bf16 per packed B-fragment tile

static constexpr size_t GATE_STRIDE = (size_t)KT64 * NT * TILE_ELEMS;  // 2,097,152 bf16

// ---------------------------------------------------------------------------
// One-time weight packing: fp32 [K=1024][N=1024] row-major -> bf16 fragment
// layout [gate][kT(0..63)][nT(0..63)][lane(0..31)][e(0..15)].
// kT < 32 -> W_h rows kT*32.., kT >= 32 -> W_x rows (kT-32)*32..
// ---------------------------------------------------------------------------
__global__ void pack_weights_kernel(
    const float* __restrict__ whi, const float* __restrict__ whf,
    const float* __restrict__ who, const float* __restrict__ whc,
    const float* __restrict__ wxi, const float* __restrict__ wxf,
    const float* __restrict__ wxo, const float* __restrict__ wxc,
    __bf16* __restrict__ packed)
{
    int idx  = blockIdx.x * blockDim.x + threadIdx.x;   // 524288 threads total
    int lane = idx & 31;
    int nT   = (idx >> 5) & 63;
    int kT   = (idx >> 11) & 63;
    int gate = idx >> 17;                               // 0..3

    const float* wh[4] = { whi, whf, who, whc };
    const float* wx[4] = { wxi, wxf, wxo, wxc };
    const float* W     = (kT < 32) ? wh[gate] : wx[gate];
    int krow0 = ((kT < 32) ? kT : (kT - 32)) * 32 + ((lane >= 16) ? 8 : 0);
    int col   = nT * 16 + (lane & 15);

    __bf16* dst = packed + ((((size_t)gate * KT64 + kT) * NT + nT) * 32 + lane) * 16;
#pragma unroll
    for (int e = 0; e < 16; ++e)
        dst[e] = (__bf16)W[(size_t)(krow0 + e + ((e >= 8) ? 8 : 0)) * HDIM + col];
}

// ---------------------------------------------------------------------------
// State init: h0 -> bf16 ping buffer, c0 -> fp32 cell buffer.
// ---------------------------------------------------------------------------
__global__ void init_state_kernel(const float* __restrict__ h0,
                                  const float* __restrict__ c0,
                                  __bf16* __restrict__ hbf,
                                  float*  __restrict__ cbuf)
{
    int i = blockIdx.x * blockDim.x + threadIdx.x;
    if (i < BH) {
        hbf[i]  = (__bf16)h0[i];
        cbuf[i] = c0[i];
    }
}

// ---------------------------------------------------------------------------
// One recurrent time step. Grid: 32 blocks x 128 threads (4 waves/block),
// 128 waves. Wave tile: 32(batch) x 16(H) x 4 gates.
//   tile = 0..127 : hT = tile & 63 (H tile), mh = tile >> 6 (batch half).
// Each wave: 64 k-steps x 4 gates x 2 batch-tiles = 512 WMMAs.
// ---------------------------------------------------------------------------
__global__ __launch_bounds__(128)
void lstm_step_kernel(const __bf16* __restrict__ packed,
                      const __bf16* __restrict__ hbf_in,
                      __bf16*       __restrict__ hbf_out,
                      const float*  __restrict__ xt,      // [64][1024] fp32 input slice
                      const float*  __restrict__ hprev,   // [64][1024] fp32 h_{t-1}
                      float*        __restrict__ cbuf,    // [64][1024] fp32 cell state
                      float*        __restrict__ out_t,   // [64][1024] fp32 h_t output
                      const float*  __restrict__ bi, const float* __restrict__ bf_,
                      const float*  __restrict__ bo, const float* __restrict__ bc,
                      const float*  __restrict__ rr)
{
    const int lane    = threadIdx.x & 31;
    const int wave    = threadIdx.x >> 5;
    const int tile    = blockIdx.x * 4 + wave;   // 0..127
    const int hT      = tile & 63;               // H tile
    const int mh      = tile >> 6;               // batch half: rows 32*mh .. 32*mh+31
    const int hi_half = (lane >= 16) ? 1 : 0;

    const int abase = hi_half ? 8 : 0;           // A k-run base within 16
    const int ncol  = hT * 16 + (lane & 15);     // C/D column (H index)
    const int arow0 = mh * 32 + (lane & 15);     // A rows for batch-tile 0
    const int arow1 = arow0 + 16;                // A rows for batch-tile 1

    // Accumulators: [batch-tile][gate], initialized with per-gate bias.
    v8f acc[2][4];
    {
        const float b0 = bi[ncol], b1 = bf_[ncol], b2 = bo[ncol], b3 = bc[ncol];
#pragma unroll
        for (int r = 0; r < 8; ++r) {
            acc[0][0][r] = b0; acc[0][1][r] = b1; acc[0][2][r] = b2; acc[0][3][r] = b3;
            acc[1][0][r] = b0; acc[1][1][r] = b1; acc[1][2][r] = b2; acc[1][3][r] = b3;
        }
    }

    const __bf16* hrow0 = hbf_in + (size_t)arow0 * HDIM;
    const __bf16* hrow1 = hbf_in + (size_t)arow1 * HDIM;
    const float*  xrow0 = xt     + (size_t)arow0 * HDIM;
    const float*  xrow1 = xt     + (size_t)arow1 * HDIM;
    const __bf16* btile = packed + ((size_t)hT * 32 + lane) * 16;  // + kT*NT*TILE_ELEMS per k-step

    // ---- K part 1: recurrent h @ W_h  (k-tiles 0..31, bf16 A from hbf_in) ----
#pragma unroll 2
    for (int ks = 0; ks < 32; ++ks) {
        const int k0 = ks * 32 + abase;
        v8bf a0lo = *(const v8bf*)(hrow0 + k0);
        v8bf a0hi = *(const v8bf*)(hrow0 + k0 + 16);
        v8bf a1lo = *(const v8bf*)(hrow1 + k0);
        v8bf a1hi = *(const v8bf*)(hrow1 + k0 + 16);
        v16bf a0 = __builtin_shufflevector(a0lo, a0hi, 0,1,2,3,4,5,6,7,8,9,10,11,12,13,14,15);
        v16bf a1 = __builtin_shufflevector(a1lo, a1hi, 0,1,2,3,4,5,6,7,8,9,10,11,12,13,14,15);

        const __bf16* bp = btile + (size_t)ks * (NT * TILE_ELEMS);
        v16bf bI = *(const v16bf*)(bp);
        v16bf bF = *(const v16bf*)(bp + GATE_STRIDE);
        v16bf bO = *(const v16bf*)(bp + 2 * GATE_STRIDE);
        v16bf bC = *(const v16bf*)(bp + 3 * GATE_STRIDE);
        __builtin_prefetch(bp + (size_t)NT * TILE_ELEMS, 0, 1);  // next k-step (global_prefetch_b8)

        acc[0][0] = __builtin_amdgcn_wmma_f32_16x16x32_bf16(false, a0, false, bI, (short)0, acc[0][0], false, false);
        acc[1][0] = __builtin_amdgcn_wmma_f32_16x16x32_bf16(false, a1, false, bI, (short)0, acc[1][0], false, false);
        acc[0][1] = __builtin_amdgcn_wmma_f32_16x16x32_bf16(false, a0, false, bF, (short)0, acc[0][1], false, false);
        acc[1][1] = __builtin_amdgcn_wmma_f32_16x16x32_bf16(false, a1, false, bF, (short)0, acc[1][1], false, false);
        acc[0][2] = __builtin_amdgcn_wmma_f32_16x16x32_bf16(false, a0, false, bO, (short)0, acc[0][2], false, false);
        acc[1][2] = __builtin_amdgcn_wmma_f32_16x16x32_bf16(false, a1, false, bO, (short)0, acc[1][2], false, false);
        acc[0][3] = __builtin_amdgcn_wmma_f32_16x16x32_bf16(false, a0, false, bC, (short)0, acc[0][3], false, false);
        acc[1][3] = __builtin_amdgcn_wmma_f32_16x16x32_bf16(false, a1, false, bC, (short)0, acc[1][3], false, false);
    }

    // ---- K part 2: input x_t @ W_x  (k-tiles 32..63, fp32 A converted inline) ----
#pragma unroll 2
    for (int ks = 0; ks < 32; ++ks) {
        const int k0 = ks * 32 + abase;
        v8f x0lo = *(const v8f*)(xrow0 + k0);
        v8f x0hi = *(const v8f*)(xrow0 + k0 + 16);
        v8f x1lo = *(const v8f*)(xrow1 + k0);
        v8f x1hi = *(const v8f*)(xrow1 + k0 + 16);
        v8bf a0lo = __builtin_convertvector(x0lo, v8bf);
        v8bf a0hi = __builtin_convertvector(x0hi, v8bf);
        v8bf a1lo = __builtin_convertvector(x1lo, v8bf);
        v8bf a1hi = __builtin_convertvector(x1hi, v8bf);
        v16bf a0 = __builtin_shufflevector(a0lo, a0hi, 0,1,2,3,4,5,6,7,8,9,10,11,12,13,14,15);
        v16bf a1 = __builtin_shufflevector(a1lo, a1hi, 0,1,2,3,4,5,6,7,8,9,10,11,12,13,14,15);

        const __bf16* bp = btile + (size_t)(32 + ks) * (NT * TILE_ELEMS);
        v16bf bI = *(const v16bf*)(bp);
        v16bf bF = *(const v16bf*)(bp + GATE_STRIDE);
        v16bf bO = *(const v16bf*)(bp + 2 * GATE_STRIDE);
        v16bf bC = *(const v16bf*)(bp + 3 * GATE_STRIDE);

        acc[0][0] = __builtin_amdgcn_wmma_f32_16x16x32_bf16(false, a0, false, bI, (short)0, acc[0][0], false, false);
        acc[1][0] = __builtin_amdgcn_wmma_f32_16x16x32_bf16(false, a1, false, bI, (short)0, acc[1][0], false, false);
        acc[0][1] = __builtin_amdgcn_wmma_f32_16x16x32_bf16(false, a0, false, bF, (short)0, acc[0][1], false, false);
        acc[1][1] = __builtin_amdgcn_wmma_f32_16x16x32_bf16(false, a1, false, bF, (short)0, acc[1][1], false, false);
        acc[0][2] = __builtin_amdgcn_wmma_f32_16x16x32_bf16(false, a0, false, bO, (short)0, acc[0][2], false, false);
        acc[1][2] = __builtin_amdgcn_wmma_f32_16x16x32_bf16(false, a1, false, bO, (short)0, acc[1][2], false, false);
        acc[0][3] = __builtin_amdgcn_wmma_f32_16x16x32_bf16(false, a0, false, bC, (short)0, acc[0][3], false, false);
        acc[1][3] = __builtin_amdgcn_wmma_f32_16x16x32_bf16(false, a1, false, bC, (short)0, acc[1][3], false, false);
    }

    // ---- Elementwise epilogue. C/D layout: VGPR r -> row r + 8*hi_half. ----
    const float rv = rr[ncol];
#pragma unroll
    for (int m = 0; m < 2; ++m) {
#pragma unroll
        for (int r = 0; r < 8; ++r) {
            const int    brow = mh * 32 + m * 16 + r + (hi_half ? 8 : 0);
            const size_t off  = (size_t)brow * HDIM + ncol;

            const float gi = acc[m][0][r], gf = acc[m][1][r];
            const float go = acc[m][2][r], gc = acc[m][3][r];
            const float it = 1.0f / (1.0f + __expf(-gi));
            const float ft = 1.0f / (1.0f + __expf(-gf));
            const float ot = 1.0f / (1.0f + __expf(-go));
            const float gt = tanhf(gc);

            const float cn = cbuf[off] * ft + it * gt;
            cbuf[off] = cn;
            const float hl = ot * tanhf(cn);
            const float hn = rv * hprev[off] + (1.0f - rv) * hl;

            out_t[off]   = hn;            // fp32 output sequence (also next hprev)
            hbf_out[off] = (__bf16)hn;    // bf16 A-matrix for next step's GEMM
        }
    }
}

// ---------------------------------------------------------------------------
// Host-side launcher (graph-capture safe: only kernel launches + MemcpyAsync).
// ---------------------------------------------------------------------------
extern "C" void kernel_launch(void* const* d_in, const int* in_sizes, int n_in,
                              void* d_out, int out_size, void* d_ws, size_t ws_size,
                              hipStream_t stream)
{
    const float* input_ = (const float*)d_in[0];
    const float* h0     = (const float*)d_in[1];
    const float* c0     = (const float*)d_in[2];
    const float* wxi    = (const float*)d_in[3];
    const float* wxf    = (const float*)d_in[4];
    const float* wxo    = (const float*)d_in[5];
    const float* wxc    = (const float*)d_in[6];
    const float* whi    = (const float*)d_in[7];
    const float* whf    = (const float*)d_in[8];
    const float* who    = (const float*)d_in[9];
    const float* whc    = (const float*)d_in[10];
    const float* bi     = (const float*)d_in[11];
    const float* bf_    = (const float*)d_in[12];
    const float* bo     = (const float*)d_in[13];
    const float* bc     = (const float*)d_in[14];
    const float* rr     = (const float*)d_in[15];
    float* out = (float*)d_out;

    // Workspace layout (~17 MB total):
    __bf16* packed = (__bf16*)d_ws;                 // 4 * GATE_STRIDE bf16 = 16 MB
    __bf16* hbfA   = packed + 4 * GATE_STRIDE;      // 128 KB
    __bf16* hbfB   = hbfA + BH;                     // 128 KB
    float*  cbuf   = (float*)(hbfB + BH);           // 256 KB

    pack_weights_kernel<<<2048, 256, 0, stream>>>(whi, whf, who, whc,
                                                  wxi, wxf, wxo, wxc, packed);
    init_state_kernel<<<BH / 256, 256, 0, stream>>>(h0, c0, hbfA, cbuf);

    for (int t = 0; t < T_STEPS; ++t) {
        const __bf16* hin  = (t & 1) ? hbfB : hbfA;
        __bf16*       hout = (t & 1) ? hbfA : hbfB;
        const float*  hprev = (t == 0) ? h0 : (out + (size_t)(t - 1) * BH);
        lstm_step_kernel<<<32, 128, 0, stream>>>(packed, hin, hout,
                                                 input_ + (size_t)t * BH, hprev,
                                                 cbuf, out + (size_t)t * BH,
                                                 bi, bf_, bo, bc, rr);
    }

    // Tuple tail: h_T == outs[T-1]; c_T from the cell buffer.
    hipMemcpyAsync(out + (size_t)T_STEPS * BH, out + (size_t)(T_STEPS - 1) * BH,
                   (size_t)BH * sizeof(float), hipMemcpyDeviceToDevice, stream);
    hipMemcpyAsync(out + (size_t)T_STEPS * BH + BH, cbuf,
                   (size_t)BH * sizeof(float), hipMemcpyDeviceToDevice, stream);
}
